// MultiScaleFeatureMapping_26474178413227
// MI455X (gfx1250) — matrix operand order repaired and checked by reading, not compiled, not run on hardware
//
#include <hip/hip_runtime.h>

// CDNA5 / gfx1250: wave32, WMMA 16x16x32 f16 -> f32 accumulate.
// GEMM mapping: M = output channels, K = (in-ch x tap) slots, N = output positions.
//
// Reduction-slot mapping chosen so B-fragment LDS gathers need only ONE
// per-lane base address plus compile-time immediate offsets:
//   conv2: slot K (0..95):  c = K & 15,  kk = 2*(K/32) + ((K%32) >> 4)
//   conv1: slot K (0..31):  c = K & 3,   kk = K >> 2
// B half h of lane L holds K%32 = 16*(L>=16) + h  ->  lds addr =
//   conv2: h*RS + (lpn + laneHi) + pp + 2*kc
//   conv1: (h&3)*RS + (lpn + 4*laneHi) + (h>>2) + pp
// Slots with kk>=5 are zeroed in the A fragment (weights); B reads there are
// in-bounds finite don't-cares.
typedef __attribute__((ext_vector_type(16))) _Float16 v16h;
typedef __attribute__((ext_vector_type(8)))  float    v8f;

#define QCH   256            // pooled positions per workgroup (8 waves x 2 tiles x 16)
#define RS    520            // LDS row stride in halves (2*QCH+8, padded)
#define UNITS (RS/2)         // u32 copy units per channel row (260)

// ---------------------------------------------------------------------------
// Kernel 0: coarse-grain (mean over windows of s) + f32->f16.
// Natural layout xs[b][c][pos]; each thread emits 8 positions (one uint4).
// ---------------------------------------------------------------------------
__global__ __launch_bounds__(256) void cg_kernel(
    const float* __restrict__ x, _Float16* __restrict__ xs, int Ls, int s)
{
  size_t gid   = (size_t)blockIdx.x * 256 + threadIdx.x;
  int    nvec  = Ls >> 3;
  size_t total = (size_t)2048 * 4 * nvec;
  if (gid >= total) return;
  int pos8 = (int)(gid % nvec);
  int rem  = (int)(gid / nvec);
  int c    = rem & 3;
  int b    = rem >> 2;
  float inv = 1.0f / (float)s;
  const float* src = x + ((size_t)(b*4 + c))*2048 + (size_t)pos8*8*s;
  union { _Float16 h[8]; uint4 u; } pack;
  #pragma unroll
  for (int j = 0; j < 8; ++j) {
    float sum = 0.f;
    for (int jj = 0; jj < s; ++jj) sum += src[j*s + jj];
    pack.h[j] = (_Float16)(sum * inv);
  }
  *(uint4*)(xs + ((size_t)(b*4 + c))*Ls + (size_t)pos8*8) = pack.u;
}

// ---------------------------------------------------------------------------
// Kernel 1: conv1 (16<-4, K=5) + bias + ReLU + maxpool2 via WMMA.
// ---------------------------------------------------------------------------
__global__ __launch_bounds__(256) void conv1_kernel(
    const _Float16* __restrict__ xs, const float* __restrict__ W1,
    const float* __restrict__ b1, _Float16* __restrict__ h1,
    int Ls, int len1)
{
  __shared__ uint tile[4*UNITS];             // 4 ch x 520 pos halves (4.2 KB)
  const int b      = blockIdx.x;
  const int qstart = blockIdx.y * QCH;
  const int pstart = 2*qstart;
  const int tid    = threadIdx.x;
  const int lane   = tid & 31;
  const int wave   = tid >> 5;
  const int laneHi = lane >> 4;
  const int n      = lane & 15;              // q offset within tile (WMMA N)

  // --- cooperative LDS fill: contiguous u32 loads along positions (pad->0)
  for (int i = tid; i < 4*UNITS; i += 256) {
    int c = i / UNITS, j = i - c*UNITS;
    int p0 = pstart + 2*j;
    const _Float16* src = xs + ((size_t)(b*4 + c))*Ls + p0;
    uint v = 0u;
    if (p0 + 1 < Ls)      v = *(const uint*)src;
    else if (p0 < Ls)     v = (uint)*(const unsigned short*)src;
    tile[c*UNITS + j] = v;
  }
  if (blockIdx.y + 1 < gridDim.y && tid < 4)   // global_prefetch_b8 for next chunk
    __builtin_prefetch(xs + ((size_t)(b*4 + tid))*Ls + pstart + RS, 0, 1);
  __syncthreads();

  // --- A fragment: slot K -> (c = K&3, kk = K>>2), zero kk>=5.
  // A layout (matches B gather): K = laneHi*16 + h, M = lane&15.
  v16h afrag;
  #pragma unroll
  for (int h = 0; h < 16; ++h) {
    int vr = h >> 1, pb = h & 1;
    int K  = h + (laneHi << 4);
    int c  = K & 3, kk = K >> 2;
    float w = W1[(lane & 15)*20 + c*5 + (kk < 5 ? kk : 0)];  // clamped load
    afrag[h] = (kk < 5) ? (_Float16)w : (_Float16)0.0f;      // cndmask, no branch
  }

  const _Float16* lds = (const _Float16*)tile;

  #pragma unroll
  for (int t = 0; t < 2; ++t) {              // two 16-q tiles per wave
    int qbase = qstart + wave*32 + t*16;
    if (qbase >= len1) continue;             // wave-uniform skip (EXEC stays all-1)
    int lpn = 2*(wave*32 + t*16 + n);
    const _Float16* base = lds + lpn + laneHi*4;   // single per-lane base
    v8f acc[2] = {};
    #pragma unroll
    for (int pp = 0; pp < 2; ++pp) {
      v16h bfrag;
      #pragma unroll
      for (int h = 0; h < 16; ++h)           // immediate-offset ds_load_u16
        bfrag[h] = base[(h & 3)*RS + (h >> 2) + pp];
      acc[pp] = __builtin_amdgcn_wmma_f32_16x16x32_f16(
          false, afrag, false, bfrag, (short)0, acc[pp], false, false);
    }
    int q = qbase + n;
    if (q < len1) {
      #pragma unroll
      for (int r = 0; r < 8; ++r) {
        int o = r + (laneHi << 3);
        float val = fmaxf(acc[0][r], acc[1][r]) + b1[o];
        val = fmaxf(val, 0.0f);
        h1[((size_t)(b*16 + o))*len1 + q] = (_Float16)val;
      }
    }
  }
}

// ---------------------------------------------------------------------------
// Kernel 2: conv2 (32<-16, K=5) + bias + ReLU + maxpool2 via WMMA.
// 2 M-tiles x 3 K-chunks x 2 pre-pool positions = 12 WMMAs per 16-q tile.
// ---------------------------------------------------------------------------
__global__ __launch_bounds__(256) void conv2_kernel(
    const _Float16* __restrict__ h1, const float* __restrict__ W2,
    const float* __restrict__ b2, float* __restrict__ out,
    int len1, int len2, int outoff, int outW)
{
  __shared__ uint tile[16*UNITS];            // 16 ch x 520 pos halves (16.6 KB)
  const int b      = blockIdx.x;
  const int qstart = blockIdx.y * QCH;
  const int pstart = 2*qstart;
  const int tid    = threadIdx.x;
  const int lane   = tid & 31;
  const int wave   = tid >> 5;
  const int laneHi = lane >> 4;
  const int n      = lane & 15;

  for (int i = tid; i < 16*UNITS; i += 256) {
    int c = i / UNITS, j = i - c*UNITS;
    int p0 = pstart + 2*j;
    const _Float16* src = h1 + ((size_t)(b*16 + c))*len1 + p0;
    uint v = 0u;
    if (p0 + 1 < len1)    v = *(const uint*)src;
    else if (p0 < len1)   v = (uint)*(const unsigned short*)src;
    tile[c*UNITS + j] = v;
  }
  if (blockIdx.y + 1 < gridDim.y && tid < 16)
    __builtin_prefetch(h1 + ((size_t)(b*16 + tid))*len1 + pstart + RS, 0, 1);
  __syncthreads();

  // --- 6 A fragments [mtile][kchunk]: slot (kc, K_local) -> (c = K&15,
  //     kk = 2*kc + (K>>4)); zero kk>=5 (i.e. kc==2 upper half).
  v16h afrag[2][3];
  #pragma unroll
  for (int mt = 0; mt < 2; ++mt) {
    #pragma unroll
    for (int kc = 0; kc < 3; ++kc) {
      #pragma unroll
      for (int h = 0; h < 16; ++h) {
        int vr = h >> 1, pb = h & 1;
        int K  = h + (laneHi << 4);
        int c  = K & 15, kk = 2*kc + (K >> 4);
        int m  = mt*16 + (lane & 15);
        float w = W2[m*80 + c*5 + (kk < 5 ? kk : 0)];
        afrag[mt][kc][h] = (kk < 5) ? (_Float16)w : (_Float16)0.0f;
      }
    }
  }

  const _Float16* lds = (const _Float16*)tile;

  #pragma unroll
  for (int t = 0; t < 2; ++t) {
    int qbase = qstart + wave*32 + t*16;
    if (qbase >= len2) continue;             // wave-uniform skip
    int lpn = 2*(wave*32 + t*16 + n);
    const _Float16* base = lds + lpn + laneHi;     // single per-lane base
    v8f acc[2][2] = {};                      // [pre-pool pos][mtile]
    #pragma unroll
    for (int pp = 0; pp < 2; ++pp) {
      #pragma unroll
      for (int kc = 0; kc < 3; ++kc) {
        v16h bfrag;
        #pragma unroll
        for (int h = 0; h < 16; ++h)         // immediate-offset ds_load_u16
          bfrag[h] = base[h*RS + pp + 2*kc];
        acc[pp][0] = __builtin_amdgcn_wmma_f32_16x16x32_f16(
            false, afrag[0][kc], false, bfrag, (short)0, acc[pp][0], false, false);
        acc[pp][1] = __builtin_amdgcn_wmma_f32_16x16x32_f16(
            false, afrag[1][kc], false, bfrag, (short)0, acc[pp][1], false, false);
      }
    }
    int q = qbase + n;
    if (q < len2) {
      #pragma unroll
      for (int mt = 0; mt < 2; ++mt) {
        #pragma unroll
        for (int r = 0; r < 8; ++r) {
          int o = mt*16 + r + (laneHi << 3);
          float val = fmaxf(acc[0][mt][r], acc[1][mt][r]) + b2[o];
          val = fmaxf(val, 0.0f);
          out[((size_t)(b*32 + o))*outW + outoff + q] = val;
        }
      }
    }
  }
}

// ---------------------------------------------------------------------------
extern "C" void kernel_launch(void* const* d_in, const int* in_sizes, int n_in,
                              void* d_out, int out_size, void* d_ws, size_t ws_size,
                              hipStream_t stream) {
  (void)in_sizes; (void)n_in; (void)out_size; (void)ws_size;
  const float* x  = (const float*)d_in[0];
  const float* W1 = (const float*)d_in[1];
  const float* b1 = (const float*)d_in[2];
  const float* W2 = (const float*)d_in[3];
  const float* b2 = (const float*)d_in[4];
  float* out = (float*)d_out;
  char*  ws  = (char*)d_ws;

  const int Ls_[3]   = {2048, 1024, 512};
  const int len1_[3] = {1022,  510, 254};   // (Ls-4)/2
  const int len2_[3] = { 509,  253, 125};   // (len1-4)/2
  const int off_[3]  = {   0,  509, 762};   // concat offsets; total width 887
  const int sc_[3]   = {   1,    2,   4};
  const int outW = 887;

  size_t xs_off[3], h1_off[3], cur = 0;
  for (int i = 0; i < 3; ++i) { xs_off[i] = cur; cur += (size_t)2048*4*Ls_[i]*2; }
  for (int i = 0; i < 3; ++i) { h1_off[i] = cur; cur += (size_t)2048*16*len1_[i]*2; }

  for (int i = 0; i < 3; ++i) {
    _Float16* xs = (_Float16*)(ws + xs_off[i]);
    size_t total = (size_t)2048*4*(Ls_[i] >> 3);
    int blocks = (int)((total + 255) / 256);
    cg_kernel<<<blocks, 256, 0, stream>>>(x, xs, Ls_[i], sc_[i]);
  }
  for (int i = 0; i < 3; ++i) {
    _Float16* xs = (_Float16*)(ws + xs_off[i]);
    _Float16* h1 = (_Float16*)(ws + h1_off[i]);
    dim3 grid(2048, (len1_[i] + QCH - 1) / QCH);
    conv1_kernel<<<grid, 256, 0, stream>>>(xs, W1, b1, h1, Ls_[i], len1_[i]);
  }
  for (int i = 0; i < 3; ++i) {
    _Float16* h1 = (_Float16*)(ws + h1_off[i]);
    dim3 grid(2048, (len2_[i] + QCH - 1) / QCH);
    conv2_kernel<<<grid, 256, 0, stream>>>(h1, W2, b2, out,
                                           len1_[i], len2_[i], off_[i], outW);
  }
}
